// Attention_74586402062900
// MI455X (gfx1250) — matrix operand order repaired and checked by reading, not compiled, not run on hardware
//
#include <hip/hip_runtime.h>
#include <hip/hip_bf16.h>
#include <math.h>

// Problem constants (from reference)
#define NT 128
#define NB 256
#define NC 512
#define NH 512
#define NCLS 37
#define NSTEPS 32

typedef __attribute__((ext_vector_type(16))) __bf16 v16bf;
typedef __attribute__((ext_vector_type(8)))  __bf16 v8bf;
typedef __attribute__((ext_vector_type(8)))  float  v8f;

#define WMMA_BF16(ACC, AF, BF)                                                 \
    (ACC) = __builtin_amdgcn_wmma_f32_16x16x32_bf16(                           \
        false, (AF), false, (BF), (short)0, (ACC), false, false)

// ---------------------------------------------------------------------------
// WMMA GEMM on pre-converted bf16 operands:
//   C[M,Nreal] = A[M,K] @ W[Nreal,K]^T (+ bias[Nreal])
// One wave computes a 16(M) x 64(N) tile: 4 independent accumulators reuse
// the A fragment 4x per K-chunk. Inner loop is pure b128 loads + 4 WMMAs
// (no conversions). N edge handled by clamping B rows (store-masked) so the
// K loop has no control flow. M must be a multiple of 16. Waves in a block
// share the same N range so B fragments hit the WGP$.
// A fragment (16x32, lane M=r): elems 0..15 at K=k0+16*half (contiguous).
// B fragment (32x16, lane N=r): elems 0..15 at K=k0+16*half (contiguous).
// ---------------------------------------------------------------------------
template <typename OutT>
__global__ void gemm_bf16_wmma(const __bf16* __restrict__ A,
                               const __bf16* __restrict__ W,
                               const float* __restrict__ bias,
                               OutT* __restrict__ C,
                               int M, int Nreal, int K, int ldc)
{
    const int wave   = threadIdx.x >> 5;
    const int tile   = blockIdx.x * (blockDim.x >> 5) + wave;
    const int tilesM = M >> 4;
    const int tilesN = (Nreal + 63) >> 6;
    if (tile >= tilesM * tilesN) return;         // wave-uniform: EXEC stays full

    const int tn   = tile / tilesM;              // block-mates share tn
    const int tm   = tile % tilesM;
    const int lane = threadIdx.x & 31;
    const int half = lane >> 4;
    const int r    = lane & 15;

    const __bf16* arow = A + (size_t)(tm * 16 + r) * (size_t)K;

    const int nb0 = tn * 64 + r;
    const int nb1 = nb0 + 16, nb2 = nb0 + 32, nb3 = nb0 + 48;
    // Clamp to a valid weight row; garbage columns are masked at the store.
    const __bf16* w0 = W + (size_t)(nb0 < Nreal ? nb0 : Nreal - 1) * (size_t)K;
    const __bf16* w1 = W + (size_t)(nb1 < Nreal ? nb1 : Nreal - 1) * (size_t)K;
    const __bf16* w2 = W + (size_t)(nb2 < Nreal ? nb2 : Nreal - 1) * (size_t)K;
    const __bf16* w3 = W + (size_t)(nb3 < Nreal ? nb3 : Nreal - 1) * (size_t)K;

    v8f acc0 = {}, acc1 = {}, acc2 = {}, acc3 = {};

    for (int k0 = 0; k0 < K; k0 += 32) {
        v16bf af = *(const v16bf*)(arow + k0 + 16 * half);
        v16bf bf0 = *(const v16bf*)(w0 + k0 + 16 * half);
        v16bf bf1 = *(const v16bf*)(w1 + k0 + 16 * half);
        v16bf bf2 = *(const v16bf*)(w2 + k0 + 16 * half);
        v16bf bf3 = *(const v16bf*)(w3 + k0 + 16 * half);
        WMMA_BF16(acc0, af, bf0);
        WMMA_BF16(acc1, af, bf1);
        WMMA_BF16(acc2, af, bf2);
        WMMA_BF16(acc3, af, bf3);
    }

    const int mbase = tm * 16 + 8 * half;        // C/D: row = vgpr + 8*half
    #pragma unroll
    for (int i = 0; i < 8; ++i) {
        OutT* crow = C + (size_t)(mbase + i) * (size_t)ldc;
        if (nb0 < Nreal) crow[nb0] = (OutT)(acc0[i] + (bias ? bias[nb0] : 0.0f));
        if (nb1 < Nreal) crow[nb1] = (OutT)(acc1[i] + (bias ? bias[nb1] : 0.0f));
        if (nb2 < Nreal) crow[nb2] = (OutT)(acc2[i] + (bias ? bias[nb2] : 0.0f));
        if (nb3 < Nreal) crow[nb3] = (OutT)(acc3[i] + (bias ? bias[nb3] : 0.0f));
    }
}

// ---------------------------------------------------------------------------
// f32 -> bf16 bulk conversion (one-shot for feats + weights).
// ---------------------------------------------------------------------------
__global__ void cvt_f32_bf16_kernel(const float* __restrict__ src,
                                    __bf16* __restrict__ dst, int n)
{
    const int i = blockIdx.x * blockDim.x + threadIdx.x;
    if (i < n) dst[i] = (__bf16)src[i];
}

// ---------------------------------------------------------------------------
// e[b*NT + t] = sum_h tanh(feats_proj[t,b,h] + hp[b,h]) * w_score[h]
// feats_proj is bf16; hp/w_score f32. One wave per (t,b) pair.
// ---------------------------------------------------------------------------
__global__ void attn_energy_kernel(const __bf16* __restrict__ feats_proj,
                                   const float* __restrict__ hp,
                                   const float* __restrict__ w_score,
                                   float* __restrict__ e)
{
    const int wave = threadIdx.x >> 5;
    const int pair = blockIdx.x * (blockDim.x >> 5) + wave;   // NT*NB pairs
    const int t = pair / NB;
    const int b = pair % NB;
    const int lane = threadIdx.x & 31;

    const __bf16* fp = feats_proj + ((size_t)t * NB + b) * NH;
    const float*  hb = hp + (size_t)b * NH;

    float acc = 0.0f;
    #pragma unroll
    for (int c = 0; c < NH / 256; ++c) {
        const int base = c * 256 + lane * 8;
        v8bf f8 = *(const v8bf*)(fp + base);
        float4 h0 = *(const float4*)(hb + base);
        float4 h1 = *(const float4*)(hb + base + 4);
        float4 w0 = *(const float4*)(w_score + base);
        float4 w1 = *(const float4*)(w_score + base + 4);
        acc += tanhf((float)f8[0] + h0.x) * w0.x;
        acc += tanhf((float)f8[1] + h0.y) * w0.y;
        acc += tanhf((float)f8[2] + h0.z) * w0.z;
        acc += tanhf((float)f8[3] + h0.w) * w0.w;
        acc += tanhf((float)f8[4] + h1.x) * w1.x;
        acc += tanhf((float)f8[5] + h1.y) * w1.y;
        acc += tanhf((float)f8[6] + h1.z) * w1.z;
        acc += tanhf((float)f8[7] + h1.w) * w1.w;
    }
    #pragma unroll
    for (int off = 16; off > 0; off >>= 1)
        acc += __shfl_xor(acc, off, 32);
    if (lane == 0)
        e[(size_t)b * NT + t] = acc;
}

// ---------------------------------------------------------------------------
// Softmax over T per batch element: alpha[b, :] = softmax(e[b, :]), T = 128.
// ---------------------------------------------------------------------------
__global__ void softmax_t_kernel(const float* __restrict__ e,
                                 float* __restrict__ alpha)
{
    const int b = blockIdx.x;
    const int t = threadIdx.x;          // 128 threads
    __shared__ float red[NT];

    const float v = e[(size_t)b * NT + t];
    red[t] = v; __syncthreads();
    for (int s = NT / 2; s > 0; s >>= 1) {
        if (t < s) red[t] = fmaxf(red[t], red[t + s]);
        __syncthreads();
    }
    const float m = red[0];
    __syncthreads();

    const float ex = expf(v - m);
    red[t] = ex; __syncthreads();
    for (int s = NT / 2; s > 0; s >>= 1) {
        if (t < s) red[t] += red[t + s];
        __syncthreads();
    }
    alpha[(size_t)b * NT + t] = ex / red[0];
}

// ---------------------------------------------------------------------------
// ctx_bf[b,c] = (bf16) sum_t feats[t,b,c] * alpha[b,t]  (feats L2-resident)
// ctx is only ever a GEMM-A operand -> produce bf16 directly.
// ---------------------------------------------------------------------------
__global__ void context_kernel(const float* __restrict__ feats,
                               const float* __restrict__ alpha,
                               __bf16* __restrict__ ctx_bf)
{
    const int c = blockIdx.x * blockDim.x + threadIdx.x;   // NC
    const int b = blockIdx.y;
    const float* fp = feats + (size_t)b * NC + c;
    const float* ab = alpha + (size_t)b * NT;
    float acc = 0.0f;
    #pragma unroll 4
    for (int t = 0; t < NT; ++t)
        acc += fp[(size_t)t * NB * NC] * ab[t];
    ctx_bf[(size_t)b * NC + c] = (__bf16)acc;
}

// ---------------------------------------------------------------------------
// GRU elementwise (PyTorch GRUCell): consumes gi/gh [B,3H]; updates f32 h,
// its bf16 mirror (next step's GEMM-A), and scatters bf16 into hs at
// [b*NSTEPS + step] (classifier-ready order).
// ---------------------------------------------------------------------------
__global__ void gru_elementwise_kernel(const float* __restrict__ gi,
                                       const float* __restrict__ gh,
                                       float* __restrict__ h,
                                       __bf16* __restrict__ h_bf,
                                       __bf16* __restrict__ hs_bf,
                                       int step)
{
    const int idx = blockIdx.x * blockDim.x + threadIdx.x;  // NB*NH
    const int b = idx / NH;
    const int j = idx % NH;
    const float* gib = gi + (size_t)b * 3 * NH;
    const float* ghb = gh + (size_t)b * 3 * NH;

    const float r = 1.0f / (1.0f + expf(-(gib[j] + ghb[j])));
    const float z = 1.0f / (1.0f + expf(-(gib[NH + j] + ghb[NH + j])));
    const float n = tanhf(gib[2 * NH + j] + r * ghb[2 * NH + j]);
    const float hnew = (1.0f - z) * n + z * h[idx];

    h[idx]    = hnew;
    h_bf[idx] = (__bf16)hnew;
    hs_bf[((size_t)b * NSTEPS + step) * NH + j] = (__bf16)hnew;
}

__global__ void init_h_kernel(float* __restrict__ h,
                              __bf16* __restrict__ h_bf, int n)
{
    const int i = blockIdx.x * blockDim.x + threadIdx.x;
    if (i < n) { h[i] = 0.0f; h_bf[i] = (__bf16)0.0f; }
}

// ---------------------------------------------------------------------------
// Host orchestration (graph-capture safe: only kernel launches on `stream`).
// ---------------------------------------------------------------------------
extern "C" void kernel_launch(void* const* d_in, const int* in_sizes, int n_in,
                              void* d_out, int out_size, void* d_ws, size_t ws_size,
                              hipStream_t stream)
{
    (void)in_sizes; (void)n_in; (void)out_size; (void)ws_size;

    const float* feats   = (const float*)d_in[0];   // [NT,NB,NC]
    // d_in[1] = text_length (constant == NSTEPS, unused)
    const float* W_i2h   = (const float*)d_in[2];   // [NH,NC]
    const float* W_h2h   = (const float*)d_in[3];   // [NH,NH]
    const float* b_h2h   = (const float*)d_in[4];   // [NH]
    const float* w_score = (const float*)d_in[5];   // [NH]
    const float* W_ih    = (const float*)d_in[6];   // [3NH,NC]
    const float* b_ih    = (const float*)d_in[7];   // [3NH]
    const float* W_hh    = (const float*)d_in[8];   // [3NH,NH]
    const float* b_hh    = (const float*)d_in[9];   // [3NH]
    const float* W_gen   = (const float*)d_in[10];  // [NCLS,NH]
    const float* b_gen   = (const float*)d_in[11];  // [NCLS]
    float* probs = (float*)d_out;                   // [NB*NSTEPS, NCLS]

    // Workspace carving (64B-aligned slabs)
    char* base = (char*)d_ws;
    auto alloc_f = [&](size_t n) { float*  p = (float*)base;  base += ((n * 4 + 63) & ~(size_t)63); return p; };
    auto alloc_b = [&](size_t n) { __bf16* p = (__bf16*)base; base += ((n * 2 + 63) & ~(size_t)63); return p; };

    float*  hp       = alloc_f((size_t)NB * NH);
    float*  e        = alloc_f((size_t)NB * NT);
    float*  alpha    = alloc_f((size_t)NB * NT);
    float*  gi       = alloc_f((size_t)NB * 3 * NH);
    float*  gh       = alloc_f((size_t)NB * 3 * NH);
    float*  h        = alloc_f((size_t)NB * NH);
    __bf16* fproj_bf = alloc_b((size_t)NT * NB * NH);   // 33.5 MB
    __bf16* feats_bf = alloc_b((size_t)NT * NB * NC);   // 33.5 MB
    __bf16* Wi2h_bf  = alloc_b((size_t)NH * NC);
    __bf16* Wh2h_bf  = alloc_b((size_t)NH * NH);
    __bf16* Wih_bf   = alloc_b((size_t)3 * NH * NC);
    __bf16* Whh_bf   = alloc_b((size_t)3 * NH * NH);
    __bf16* Wgen_bf  = alloc_b((size_t)NCLS * NH);
    __bf16* h_bf     = alloc_b((size_t)NB * NH);
    __bf16* ctx_bf   = alloc_b((size_t)NB * NC);
    __bf16* hs_bf    = alloc_b((size_t)NSTEPS * NB * NH); // 8.4 MB

    auto cvt = [&](const float* src, __bf16* dst, int n) {
        cvt_f32_bf16_kernel<<<(n + 255) / 256, 256, 0, stream>>>(src, dst, n);
    };
    auto gemm_f32out = [&](const __bf16* A, const __bf16* W, const float* bias,
                           float* C, int M, int Nreal, int K, int ldc) {
        const int tiles  = (M / 16) * ((Nreal + 63) / 64);
        gemm_bf16_wmma<float><<<(tiles + 7) / 8, 256, 0, stream>>>(A, W, bias, C, M, Nreal, K, ldc);
    };
    auto gemm_bfout = [&](const __bf16* A, const __bf16* W, const float* bias,
                          __bf16* C, int M, int Nreal, int K, int ldc) {
        const int tiles  = (M / 16) * ((Nreal + 63) / 64);
        gemm_bf16_wmma<__bf16><<<(tiles + 7) / 8, 256, 0, stream>>>(A, W, bias, C, M, Nreal, K, ldc);
    };

    // One-shot conversions to bf16
    cvt(feats, feats_bf, NT * NB * NC);
    cvt(W_i2h, Wi2h_bf, NH * NC);
    cvt(W_h2h, Wh2h_bf, NH * NH);
    cvt(W_ih,  Wih_bf,  3 * NH * NC);
    cvt(W_hh,  Whh_bf,  3 * NH * NH);
    cvt(W_gen, Wgen_bf, NCLS * NH);

    // h0 = 0
    init_h_kernel<<<(NB * NH + 255) / 256, 256, 0, stream>>>(h, h_bf, NB * NH);

    // feats_proj[t,b,:] = feats[t,b,:] @ W_i2h^T  (one-shot; bf16 output)
    gemm_bfout(feats_bf, Wi2h_bf, nullptr, fproj_bf, NT * NB, NH, NC, NH);

    for (int step = 0; step < NSTEPS; ++step) {
        // hp = h @ W_h2h^T + b_h2h
        gemm_f32out(h_bf, Wh2h_bf, b_h2h, hp, NB, NH, NH, NH);
        // e[b,t] = tanh(feats_proj + hp) . w_score
        attn_energy_kernel<<<(NT * NB) / 8, 256, 0, stream>>>(fproj_bf, hp, w_score, e);
        // alpha = softmax over T
        softmax_t_kernel<<<NB, NT, 0, stream>>>(e, alpha);
        // context[b,c] = sum_t feats[t,b,c] * alpha[b,t]  (bf16 out)
        context_kernel<<<dim3(NC / 256, NB), 256, 0, stream>>>(feats, alpha, ctx_bf);
        // GRU gates
        gemm_f32out(ctx_bf, Wih_bf, b_ih, gi, NB, 3 * NH, NC, 3 * NH);
        gemm_f32out(h_bf,   Whh_bf, b_hh, gh, NB, 3 * NH, NH, 3 * NH);
        // h update + bf16 mirrors (hs already in [b, step, H] order)
        gru_elementwise_kernel<<<(NB * NH) / 256, 256, 0, stream>>>(gi, gh, h, h_bf, hs_bf, step);
    }

    // probs = hs @ W_gen^T + b_gen   (N=37 masked within a single 64-wide tile)
    gemm_f32out(hs_bf, Wgen_bf, b_gen, probs, NB * NSTEPS, NCLS, NH, NCLS);
}
